// TrajectoryPredictor_71176198029848
// MI455X (gfx1250) — matrix operand order repaired and checked by reading, not compiled, not run on hardware
//
#include <hip/hip_runtime.h>
#include <hip/hip_bf16.h>

// ---------------------------------------------------------------------------
// Model constants (from the reference)
// ---------------------------------------------------------------------------
#define BB    64
#define LL    100
#define DD    256
#define HH    4
#define DK    64          // DD / HH
#define KSZ   3
#define NSC   3
#define NLOC  10000
#define KPAD  10016       // NLOC padded up to a multiple of 32 (WMMA K step)
#define FEAT  352         // D + D/8 + 4*(D/16)
#define RR    (BB * LL)   // 6400 rows
#define MC    384         // spatial-attention row chunk (logits chunk ~15MB -> L2 resident)

typedef _Float16 f16;
typedef __attribute__((ext_vector_type(16))) _Float16 v16h;
typedef __attribute__((ext_vector_type(8)))  float    v8f;

union Frag16 { v16h v; f16 e[16]; };

// ---------------------------------------------------------------------------
// LDS-tiled WMMA GEMM:
//   C[M,N] = A[M,K](f16,row-major) * W[N,K]^T(f16,row-major) + bias[N] (opt)
// Block = 256 threads (8 waves), tile = 64(M) x 128(N), K step = 32.
// Requirements: M%64==0, K%32==0. N is generic (guarded tail).
// Each wave: 16 rows x 64 cols = 4 accumulators, A-fragment reused 4x.
// ---------------------------------------------------------------------------
#define MBLK 64
#define NBLK 128
#define KSTEP 32

__global__ __launch_bounds__(256) void gemm_f16_wmma_tiled(
    const f16* __restrict__ A, const f16* __restrict__ W,
    const float* __restrict__ bias, float* __restrict__ C,
    int M, int N, int K)
{
  __shared__ __align__(16) f16 ldsA[MBLK * KSTEP];   // 4 KB
  __shared__ __align__(16) f16 ldsB[NBLK * KSTEP];   // 8 KB

  const int tid  = threadIdx.x;
  const int lane = tid & 31;
  const int wave = tid >> 5;          // 0..7
  const int wr   = wave >> 1;         // 0..3 : 16-row group
  const int wc   = wave & 1;          // 0..1 : 64-col group

  const int row0 = blockIdx.y * MBLK;
  const int col0 = blockIdx.x * NBLK;

  // cooperative tile-load indices (8 halves/thread for A, 16 for B)
  const int a_row = (tid * 8) >> 5;         // 0..63
  const int a_ko  = (tid * 8) & 31;         // 0,8,16,24
  const int b_col = (tid * 16) >> 5;        // 0..127
  const int b_ko  = (tid * 16) & 31;        // 0,16

  // WMMA fragment lane mapping (ISA 16-bit A 16x32 / B 32x16 layouts)
  const int frow  = lane & 15;
  const int faoff = (lane >> 4) << 3;       // 0 / 8
  const int fcol  = lane & 15;
  const int fboff = (lane >> 4) << 4;       // 0 / 16

  v8f acc[4] = {};

  const f16* __restrict__ Asrc = A + (size_t)(row0 + a_row) * K + a_ko;
  const int  gcol   = col0 + b_col;
  const bool bvalid = (gcol < N);
  const f16* __restrict__ Wsrc = W + (size_t)(bvalid ? gcol : 0) * K + b_ko;

  for (int k0 = 0; k0 < K; k0 += KSTEP) {
    __syncthreads();
    *(uint4*)&ldsA[a_row * KSTEP + a_ko] = *(const uint4*)&Asrc[k0];
    uint4 b0 = make_uint4(0, 0, 0, 0), b1 = make_uint4(0, 0, 0, 0);
    if (bvalid) {
      b0 = *(const uint4*)&Wsrc[k0];
      b1 = *(const uint4*)&Wsrc[k0 + 8];
    }
    *(uint4*)&ldsB[b_col * KSTEP + b_ko]     = b0;
    *(uint4*)&ldsB[b_col * KSTEP + b_ko + 8] = b1;
    if (k0 + KSTEP < K) {                 // gfx1250 global_prefetch_b8
      __builtin_prefetch(&Asrc[k0 + KSTEP], 0, 3);
      if (bvalid) __builtin_prefetch(&Wsrc[k0 + KSTEP], 0, 3);
    }
    __syncthreads();

    Frag16 a;
#pragma unroll
    for (int e = 0; e < 8; ++e) a.e[e]     = ldsA[(wr * 16 + frow) * KSTEP + faoff + e];
#pragma unroll
    for (int e = 0; e < 8; ++e) a.e[8 + e] = ldsA[(wr * 16 + frow) * KSTEP + 16 + faoff + e];

#pragma unroll
    for (int cg = 0; cg < 4; ++cg) {
      Frag16 b;
#pragma unroll
      for (int e = 0; e < 16; ++e)
        b.e[e] = ldsB[(wc * 64 + cg * 16 + fcol) * KSTEP + (fboff >> 1) + (e < 8 ? e : 8 + e)];
      acc[cg] = __builtin_amdgcn_wmma_f32_16x16x32_f16(
          false, a.v, false, b.v, (short)0, acc[cg], false, false);
    }
  }

  // D layout: lanes 0-15 -> N=lane, VGPR v -> M=v ; lanes 16-31 -> M=8+v
  const int n     = lane & 15;
  const int mbase = (lane >> 4) << 3;
#pragma unroll
  for (int cg = 0; cg < 4; ++cg) {
    const int col = col0 + wc * 64 + cg * 16 + n;
    if (col < N) {
      const float bv = bias ? bias[col] : 0.0f;
#pragma unroll
      for (int v = 0; v < 8; ++v)
        C[(size_t)(row0 + wr * 16 + mbase + v) * N + col] = acc[cg][v] + bv;
    }
  }
}

// ---------------------------------------------------------------------------
// Elementwise / staging kernels
// ---------------------------------------------------------------------------
__global__ void cvt_f32_to_f16(const float* __restrict__ s, f16* __restrict__ d, int n) {
  int i = blockIdx.x * blockDim.x + threadIdx.x;
  if (i < n) d[i] = (f16)s[i];
}

__global__ void add_vec(const float* __restrict__ a, const float* __restrict__ b,
                        float* __restrict__ c, int n) {
  int i = blockIdx.x * blockDim.x + threadIdx.x;
  if (i < n) c[i] = a[i] + b[i];
}

__device__ __forceinline__ float gelu_exact(float x) {
  return 0.5f * x * (1.0f + erff(x * 0.70710678118654752f));
}

__global__ void gelu_vec(float* __restrict__ x, int n) {
  int i = blockIdx.x * blockDim.x + threadIdx.x;
  if (i < n) x[i] = gelu_exact(x[i]);
}

__global__ void add_loc_bias(float* __restrict__ out, const float* __restrict__ lb, int n) {
  int i = blockIdx.x * blockDim.x + threadIdx.x;
  if (i < n) out[i] += lb[i % NLOC];
}

// LayerNorm over rows of length 256 (blockDim == 256), optional exact GELU
__global__ void ln_kernel(const float* __restrict__ src, float* __restrict__ dst,
                          const float* __restrict__ g, const float* __restrict__ bta,
                          int do_gelu) {
  __shared__ float red[DD];
  const int r = blockIdx.x, tid = threadIdx.x;
  float v = src[(size_t)r * DD + tid];
  red[tid] = v; __syncthreads();
  for (int off = 128; off > 0; off >>= 1) {
    if (tid < off) red[tid] += red[tid + off];
    __syncthreads();
  }
  float mean = red[0] * (1.0f / DD); __syncthreads();
  float d = v - mean;
  red[tid] = d * d; __syncthreads();
  for (int off = 128; off > 0; off >>= 1) {
    if (tid < off) red[tid] += red[tid + off];
    __syncthreads();
  }
  float var = red[0] * (1.0f / DD);
  float y = d * rsqrtf(var + 1e-5f) * g[tid] + bta[tid];
  if (do_gelu) y = gelu_exact(y);
  dst[(size_t)r * DD + tid] = y;
}

// ---------------------------------------------------------------------------
// Embedding concat -> f16 staging (R x 352)
// ---------------------------------------------------------------------------
__global__ void embed_kernel(const int* __restrict__ loc, const int* __restrict__ usr,
                             const int* __restrict__ wk, const int* __restrict__ smin,
                             const float* __restrict__ dur, const int* __restrict__ td,
                             const float* __restrict__ loc_emb, const float* __restrict__ user_emb,
                             const float* __restrict__ wk_emb, const float* __restrict__ hour_emb,
                             const float* __restrict__ td_emb, const float* __restrict__ durW,
                             const float* __restrict__ durB, const float* __restrict__ pos_emb,
                             f16* __restrict__ out) {
  const int r = blockIdx.x;       // 0..RR-1
  const int l = r % LL;
  const int tid = threadIdx.x;    // 0..383
  if (tid >= FEAT) return;
  float v;
  if (tid < 256) {
    v = loc_emb[(size_t)loc[r] * DD + tid] + pos_emb[l * DD + tid];
  } else if (tid < 288) {
    v = user_emb[usr[r] * 32 + (tid - 256)];
  } else if (tid < 304) {
    v = wk_emb[wk[r] * 16 + (tid - 288)];
  } else if (tid < 320) {
    int hh = smin[r] / 60; hh = hh < 0 ? 0 : (hh > 24 ? 24 : hh);
    v = hour_emb[hh * 16 + (tid - 304)];
  } else if (tid < 336) {
    int t = td[r]; t = t < 0 ? 0 : (t > 100 ? 100 : t);
    v = td_emb[t * 16 + (tid - 320)];
  } else {
    v = durW[tid - 336] * dur[r] + durB[tid - 336];
  }
  out[(size_t)r * FEAT + tid] = (f16)v;
}

// ---------------------------------------------------------------------------
// Causal dilated conv support: weight repack (O,I,K)->(O, k*256+i) and im2col
// ---------------------------------------------------------------------------
__global__ void conv_w_repack(const float* __restrict__ W, f16* __restrict__ Wp, int n) {
  int idx = blockIdx.x * blockDim.x + threadIdx.x;   // over 256*768
  if (idx >= n) return;
  int o = idx / (KSZ * DD);
  int rem = idx % (KSZ * DD);
  int k = rem / DD;
  int i = rem % DD;
  Wp[idx] = (f16)W[((size_t)o * DD + i) * KSZ + k];
}

__global__ void im2col_kernel(const float* __restrict__ x, f16* __restrict__ A,
                              int dil, int n) {
  int idx = blockIdx.x * blockDim.x + threadIdx.x;   // over RR*768
  if (idx >= n) return;
  int r = idx / (KSZ * DD);
  int j = idx % (KSZ * DD);
  int k = j / DD;
  int c = j % DD;
  int b = r / LL, l = r % LL;
  int lp = l - (KSZ - 1 - k) * dil;
  float v = (lp >= 0) ? x[((size_t)(b * LL + lp)) * DD + c] : 0.0f;
  A[idx] = (f16)v;
}

// scales[r, i*256 + c] = tcb_out[r,c] + x[r,c]   (identity residual + concat)
__global__ void concat_residual(const float* __restrict__ tcb_out,
                                const float* __restrict__ x,
                                float* __restrict__ scales, int which, int n) {
  int idx = blockIdx.x * blockDim.x + threadIdx.x;   // over RR*256
  if (idx >= n) return;
  int r = idx / DD, c = idx % DD;
  scales[(size_t)r * (NSC * DD) + which * DD + c] = tcb_out[idx] + x[idx];
}

// ---------------------------------------------------------------------------
// Spatial transition staging (WMMA formulation, chunked over rows)
// ---------------------------------------------------------------------------
__global__ void tq_gather_f16(const int* __restrict__ loc, const float* __restrict__ tq_emb,
                              f16* __restrict__ tqh, int n) {
  int idx = blockIdx.x * blockDim.x + threadIdx.x;   // over RR*64
  if (idx >= n) return;
  int r = idx / DK, d = idx % DK;
  tqh[idx] = (f16)tq_emb[(size_t)loc[r] * DK + d];
}

// tkT[d, j] = tk[j, d] for j<NLOC, 0 for NLOC<=j<KPAD   (64 x KPAD, f16)
__global__ void tk_transpose_pad(const float* __restrict__ tk, f16* __restrict__ tkT, int n) {
  int idx = blockIdx.x * blockDim.x + threadIdx.x;   // over 64*KPAD
  if (idx >= n) return;
  int d = idx / KPAD, j = idx % KPAD;
  tkT[idx] = (j < NLOC) ? (f16)tk[(size_t)j * DK + d] : (f16)0.0f;
}

// softmax along a row of `ncols` f32 logits -> f16 row of `npad` (zero padded)
__global__ void row_softmax_f16pad(const float* __restrict__ logits,
                                   f16* __restrict__ w, int ncols, int npad) {
  __shared__ float red[256];
  const int r = blockIdx.x, tid = threadIdx.x;
  const float* row = logits + (size_t)r * ncols;
  float m = -1e30f;
  for (int j = tid; j < ncols; j += 256) m = fmaxf(m, row[j]);
  red[tid] = m; __syncthreads();
  for (int off = 128; off > 0; off >>= 1) {
    if (tid < off) red[tid] = fmaxf(red[tid], red[tid + off]);
    __syncthreads();
  }
  m = red[0]; __syncthreads();
  float s = 0.0f;
  for (int j = tid; j < ncols; j += 256) s += __expf(row[j] - m);
  red[tid] = s; __syncthreads();
  for (int off = 128; off > 0; off >>= 1) {
    if (tid < off) red[tid] += red[tid + off];
    __syncthreads();
  }
  const float inv = 1.0f / red[0];
  f16* wrow = w + (size_t)r * npad;
  for (int j = tid; j < npad; j += 256)
    wrow[j] = (j < ncols) ? (f16)(__expf(row[j] - m) * inv) : (f16)0.0f;
}

// ---------------------------------------------------------------------------
// Linear attention pieces
// ---------------------------------------------------------------------------
__global__ void q_head_softmax(float* __restrict__ Q) {
  __shared__ float red[DD];
  const int r = blockIdx.x, tid = threadIdx.x;
  const int s0 = tid & 63;
  const int base = tid & ~63;
  float v = Q[(size_t)r * DD + tid];
  red[tid] = v; __syncthreads();
  for (int off = 32; off > 0; off >>= 1) {
    if (s0 < off) red[tid] = fmaxf(red[tid], red[tid + off]);
    __syncthreads();
  }
  float m = red[base]; __syncthreads();
  float e = __expf(v - m);
  red[tid] = e; __syncthreads();
  for (int off = 32; off > 0; off >>= 1) {
    if (s0 < off) red[tid] += red[tid + off];
    __syncthreads();
  }
  float s = red[base];
  Q[(size_t)r * DD + tid] = e / s;
}

__global__ void k_time_softmax(float* __restrict__ K, const float* __restrict__ mask) {
  __shared__ float red[128];
  const int c = blockIdx.x, b = blockIdx.y, tid = threadIdx.x;
  float v = (tid < LL) ? K[(size_t)(b * LL + tid) * DD + c] : -1e30f;
  red[tid] = v; __syncthreads();
  for (int off = 64; off > 0; off >>= 1) {
    if (tid < off) red[tid] = fmaxf(red[tid], red[tid + off]);
    __syncthreads();
  }
  float m = red[0]; __syncthreads();
  float e = (tid < LL) ? __expf(v - m) : 0.0f;
  red[tid] = e; __syncthreads();
  for (int off = 64; off > 0; off >>= 1) {
    if (tid < off) red[tid] += red[tid + off];
    __syncthreads();
  }
  float s = red[0];
  if (tid < LL) K[(size_t)(b * LL + tid) * DD + c] = (e / s) * mask[b * LL + tid];
}

__global__ void kv_einsum(const float* __restrict__ K, const float* __restrict__ V,
                          float* __restrict__ KV) {
  const int bh = blockIdx.x;
  const int b = bh / HH, h = bh % HH;
  const int tid = threadIdx.x;
  for (int idx = tid; idx < DK * DK; idx += 256) {
    int d = idx >> 6, e = idx & 63;
    float s = 0.0f;
    for (int l = 0; l < LL; ++l) {
      size_t row = (size_t)(b * LL + l) * DD + h * DK;
      s += K[row + d] * V[row + e];
    }
    KV[(size_t)bh * (DK * DK) + idx] = s;
  }
}

__global__ void att_einsum(const float* __restrict__ Q, const float* __restrict__ KV,
                           float* __restrict__ att) {
  __shared__ float kvs[DK * DK];
  const int bh = blockIdx.x;
  const int b = bh / HH, h = bh % HH;
  const int tid = threadIdx.x;
  for (int i = tid; i < DK * DK; i += 256) kvs[i] = KV[(size_t)bh * (DK * DK) + i];
  __syncthreads();
  for (int idx = tid; idx < LL * DK; idx += 256) {
    int l = idx >> 6, e = idx & 63;
    const float* qrow = Q + (size_t)(b * LL + l) * DD + h * DK;
    float s = 0.0f;
#pragma unroll 8
    for (int d = 0; d < DK; ++d) s += qrow[d] * kvs[(d << 6) + e];
    att[(size_t)(b * LL + l) * DD + h * DK + e] = s;
  }
}

// ---------------------------------------------------------------------------
// Head: last-position select and staging
// ---------------------------------------------------------------------------
__global__ void last_pos(const float* __restrict__ mask, int* __restrict__ lp) {
  __shared__ float red[128];
  const int b = blockIdx.x, tid = threadIdx.x;
  red[tid] = (tid < LL) ? mask[b * LL + tid] : 0.0f;
  __syncthreads();
  for (int off = 64; off > 0; off >>= 1) {
    if (tid < off) red[tid] += red[tid + off];
    __syncthreads();
  }
  if (tid == 0) lp[b] = (int)(red[0] + 0.5f) - 1;
}

__global__ void gather_last(const float* __restrict__ x, const int* __restrict__ lp,
                            f16* __restrict__ lasth) {
  const int b = blockIdx.x, c = threadIdx.x;
  lasth[(size_t)b * DD + c] = (f16)x[(size_t)(b * LL + lp[b]) * DD + c];
}

// ---------------------------------------------------------------------------
// Host orchestration
// ---------------------------------------------------------------------------
static inline int cdiv(int a, int b) { return (a + b - 1) / b; }

extern "C" void kernel_launch(void* const* d_in, const int* in_sizes, int n_in,
                              void* d_out, int out_size, void* d_ws, size_t ws_size,
                              hipStream_t stream) {
  const int*   loc      = (const int*)  d_in[0];
  const int*   usr      = (const int*)  d_in[1];
  const int*   wkd      = (const int*)  d_in[2];
  const int*   smin     = (const int*)  d_in[3];
  const float* dur      = (const float*)d_in[4];
  const int*   tdiff    = (const int*)  d_in[5];
  const float* mask     = (const float*)d_in[6];
  const float* loc_emb  = (const float*)d_in[7];
  const float* user_emb = (const float*)d_in[8];
  const float* wk_emb   = (const float*)d_in[9];
  const float* hour_emb = (const float*)d_in[10];
  const float* td_emb   = (const float*)d_in[11];
  const float* dur_W    = (const float*)d_in[12];
  const float* dur_b    = (const float*)d_in[13];
  const float* pos_emb  = (const float*)d_in[14];
  const float* ff_W     = (const float*)d_in[15];
  const float* ff_b     = (const float*)d_in[16];
  const float* ff_g     = (const float*)d_in[17];
  const float* ff_beta  = (const float*)d_in[18];
  // tcb[i]: c1_W c1_b c2_W c2_b n1_g n1_b n2_g n2_b  at 19 + 8*i
  const float* fus_W    = (const float*)d_in[43];
  const float* fus_b    = (const float*)d_in[44];
  const float* ms_g     = (const float*)d_in[45];
  const float* ms_b     = (const float*)d_in[46];
  const float* tk_emb   = (const float*)d_in[47];
  const float* tq_emb   = (const float*)d_in[48];
  const float* sp_W     = (const float*)d_in[49];
  const float* sp_b     = (const float*)d_in[50];
  const float* q_W      = (const float*)d_in[51];
  const float* q_b      = (const float*)d_in[52];
  const float* k_W      = (const float*)d_in[53];
  const float* k_b      = (const float*)d_in[54];
  const float* v_W      = (const float*)d_in[55];
  const float* v_b      = (const float*)d_in[56];
  const float* o_W      = (const float*)d_in[57];
  const float* o_b      = (const float*)d_in[58];
  const float* norm_g   = (const float*)d_in[59];
  const float* norm_b   = (const float*)d_in[60];
  const float* oh_W     = (const float*)d_in[61];
  const float* oh_b     = (const float*)d_in[62];
  const float* ol_W     = (const float*)d_in[63];
  const float* ol_b     = (const float*)d_in[64];
  const float* loc_bias = (const float*)d_in[65];
  (void)in_sizes; (void)n_in; (void)out_size; (void)ws_size;

  float* out = (float*)d_out;

  // ---- workspace layout (float units)
  float* wsf = (float*)d_ws;
  size_t o = 0;
  auto alloc = [&](size_t nf) { float* p = wsf + o; o += (nf + 3) & ~(size_t)3; return p; };

  f16*   combined_h = (f16*)  alloc((size_t)RR * FEAT / 2);
  f16*   xh         = (f16*)  alloc((size_t)RR * (NSC * DD) / 2);
  f16*   wh         = (f16*)  alloc((size_t)NLOC * 128 / 2);
  float* x          =          alloc((size_t)RR * DD);
  float* tmp        =          alloc((size_t)RR * DD);
  float* h1         =          alloc((size_t)RR * DD);
  float* scales     =          alloc((size_t)RR * NSC * DD);
  float* Qf         =          alloc((size_t)RR * DD);
  float* Kf         =          alloc((size_t)RR * DD);
  float* Vf         =          alloc((size_t)RR * DD);
  float* KVm        =          alloc((size_t)BB * HH * DK * DK);
  float* attb       =          alloc((size_t)RR * DD);
  f16*   tqh        = (f16*)  alloc((size_t)RR * DK / 2);
  f16*   tkh        = (f16*)  alloc((size_t)NLOC * DK / 2);
  f16*   tkTh       = (f16*)  alloc((size_t)DK * KPAD / 2);
  float* logitc     =          alloc((size_t)MC * NLOC);
  f16*   wlh        = (f16*)  alloc((size_t)MC * KPAD / 2);
  float* spout      =          alloc((size_t)RR * DK);
  int*   lp         = (int*)   alloc(64);
  f16*   lasth      = (f16*)   alloc((size_t)BB * DD / 2);
  float* head       =          alloc((size_t)BB * 128);
  f16*   headh      = (f16*)   alloc((size_t)BB * 128 / 2);

  const dim3 blk256(256);
  auto egrid = [](int n) { return dim3(cdiv(n, 256)); };
  auto ggrid = [](int M, int N) { return dim3(cdiv(N, NBLK), M / MBLK); };

  // ---- 1. embeddings -> f16 staging (R x 352)
  embed_kernel<<<dim3(RR), dim3(384), 0, stream>>>(
      loc, usr, wkd, smin, dur, tdiff, loc_emb, user_emb, wk_emb, hour_emb,
      td_emb, dur_W, dur_b, pos_emb, combined_h);

  // ---- 2. ff: (R,352)@(256,352)^T + b -> LN -> GELU -> x
  cvt_f32_to_f16<<<egrid(DD * FEAT), blk256, 0, stream>>>(ff_W, wh, DD * FEAT);
  gemm_f16_wmma_tiled<<<ggrid(RR, DD), blk256, 0, stream>>>(combined_h, wh, ff_b, tmp, RR, DD, FEAT);
  ln_kernel<<<dim3(RR), dim3(DD), 0, stream>>>(tmp, x, ff_g, ff_beta, 1);

  // ---- 3. multi-scale dilated temporal conv blocks (all read the same x)
  for (int i = 0; i < NSC; ++i) {
    const int dil = 1 << i;
    const float* c1W = (const float*)d_in[19 + 8 * i + 0];
    const float* c1b = (const float*)d_in[19 + 8 * i + 1];
    const float* c2W = (const float*)d_in[19 + 8 * i + 2];
    const float* c2b = (const float*)d_in[19 + 8 * i + 3];
    const float* n1g = (const float*)d_in[19 + 8 * i + 4];
    const float* n1b = (const float*)d_in[19 + 8 * i + 5];
    const float* n2g = (const float*)d_in[19 + 8 * i + 6];
    const float* n2b = (const float*)d_in[19 + 8 * i + 7];

    conv_w_repack<<<egrid(DD * KSZ * DD), blk256, 0, stream>>>(c1W, wh, DD * KSZ * DD);
    im2col_kernel<<<egrid(RR * KSZ * DD), blk256, 0, stream>>>(x, xh, dil, RR * KSZ * DD);
    gemm_f16_wmma_tiled<<<ggrid(RR, DD), blk256, 0, stream>>>(xh, wh, c1b, tmp, RR, DD, KSZ * DD);
    ln_kernel<<<dim3(RR), dim3(DD), 0, stream>>>(tmp, h1, n1g, n1b, 1);

    conv_w_repack<<<egrid(DD * KSZ * DD), blk256, 0, stream>>>(c2W, wh, DD * KSZ * DD);
    im2col_kernel<<<egrid(RR * KSZ * DD), blk256, 0, stream>>>(h1, xh, dil, RR * KSZ * DD);
    gemm_f16_wmma_tiled<<<ggrid(RR, DD), blk256, 0, stream>>>(xh, wh, c2b, tmp, RR, DD, KSZ * DD);
    ln_kernel<<<dim3(RR), dim3(DD), 0, stream>>>(tmp, h1, n2g, n2b, 1);

    concat_residual<<<egrid(RR * DD), blk256, 0, stream>>>(h1, x, scales, i, RR * DD);
  }

  // ---- 4. fusion: (R,768)@(256,768)^T -> LN -> x
  cvt_f32_to_f16<<<egrid(RR * NSC * DD), blk256, 0, stream>>>(scales, xh, RR * NSC * DD);
  cvt_f32_to_f16<<<egrid(DD * NSC * DD), blk256, 0, stream>>>(fus_W, wh, DD * NSC * DD);
  gemm_f16_wmma_tiled<<<ggrid(RR, DD), blk256, 0, stream>>>(xh, wh, fus_b, tmp, RR, DD, NSC * DD);
  ln_kernel<<<dim3(RR), dim3(DD), 0, stream>>>(tmp, x, ms_g, ms_b, 0);

  // ---- 5. spatial transition on WMMA, chunked so logits stay L2-resident
  tq_gather_f16<<<egrid(RR * DK), blk256, 0, stream>>>(loc, tq_emb, tqh, RR * DK);
  cvt_f32_to_f16<<<egrid(NLOC * DK), blk256, 0, stream>>>(tk_emb, tkh, NLOC * DK);
  tk_transpose_pad<<<egrid(DK * KPAD), blk256, 0, stream>>>(tk_emb, tkTh, DK * KPAD);
  for (int m0 = 0; m0 < RR; m0 += MC) {
    int Mc = RR - m0; if (Mc > MC) Mc = MC;
    // logits = tq @ tk^T   (Mc x NLOC)
    gemm_f16_wmma_tiled<<<ggrid(Mc, NLOC), blk256, 0, stream>>>(
        tqh + (size_t)m0 * DK, tkh, nullptr, logitc, Mc, NLOC, DK);
    // row softmax -> f16, K-padded
    row_softmax_f16pad<<<dim3(Mc), blk256, 0, stream>>>(logitc, wlh, NLOC, KPAD);
    // spout = w @ tk       (Mc x 64)
    gemm_f16_wmma_tiled<<<ggrid(Mc, DK), blk256, 0, stream>>>(
        wlh, tkTh, nullptr, spout + (size_t)m0 * DK, Mc, DK, KPAD);
  }
  cvt_f32_to_f16<<<egrid(RR * DK), blk256, 0, stream>>>(spout, xh, RR * DK);
  cvt_f32_to_f16<<<egrid(DD * DK), blk256, 0, stream>>>(sp_W, wh, DD * DK);
  gemm_f16_wmma_tiled<<<ggrid(RR, DD), blk256, 0, stream>>>(xh, wh, sp_b, tmp, RR, DD, DK);
  add_vec<<<egrid(RR * DD), blk256, 0, stream>>>(x, tmp, x, RR * DD);

  // ---- 6. linear attention
  cvt_f32_to_f16<<<egrid(RR * DD), blk256, 0, stream>>>(x, xh, RR * DD);

  cvt_f32_to_f16<<<egrid(DD * DD), blk256, 0, stream>>>(q_W, wh, DD * DD);
  gemm_f16_wmma_tiled<<<ggrid(RR, DD), blk256, 0, stream>>>(xh, wh, q_b, Qf, RR, DD, DD);
  q_head_softmax<<<dim3(RR), dim3(DD), 0, stream>>>(Qf);

  cvt_f32_to_f16<<<egrid(DD * DD), blk256, 0, stream>>>(k_W, wh, DD * DD);
  gemm_f16_wmma_tiled<<<ggrid(RR, DD), blk256, 0, stream>>>(xh, wh, k_b, Kf, RR, DD, DD);
  k_time_softmax<<<dim3(DD, BB), dim3(128), 0, stream>>>(Kf, mask);

  cvt_f32_to_f16<<<egrid(DD * DD), blk256, 0, stream>>>(v_W, wh, DD * DD);
  gemm_f16_wmma_tiled<<<ggrid(RR, DD), blk256, 0, stream>>>(xh, wh, v_b, Vf, RR, DD, DD);

  kv_einsum<<<dim3(BB * HH), blk256, 0, stream>>>(Kf, Vf, KVm);
  att_einsum<<<dim3(BB * HH), blk256, 0, stream>>>(Qf, KVm, attb);

  cvt_f32_to_f16<<<egrid(RR * DD), blk256, 0, stream>>>(attb, xh, RR * DD);
  cvt_f32_to_f16<<<egrid(DD * DD), blk256, 0, stream>>>(o_W, wh, DD * DD);
  gemm_f16_wmma_tiled<<<ggrid(RR, DD), blk256, 0, stream>>>(xh, wh, o_b, tmp, RR, DD, DD);
  ln_kernel<<<dim3(RR), dim3(DD), 0, stream>>>(tmp, x, norm_g, norm_b, 0);

  // ---- 7. head: last position -> 128 GELU -> 10000 logits
  last_pos<<<dim3(BB), dim3(128), 0, stream>>>(mask, lp);
  gather_last<<<dim3(BB), dim3(DD), 0, stream>>>(x, lp, lasth);

  cvt_f32_to_f16<<<egrid(128 * DD), blk256, 0, stream>>>(oh_W, wh, 128 * DD);
  gemm_f16_wmma_tiled<<<ggrid(BB, 128), blk256, 0, stream>>>(lasth, wh, oh_b, head, BB, 128, DD);
  gelu_vec<<<egrid(BB * 128), blk256, 0, stream>>>(head, BB * 128);
  cvt_f32_to_f16<<<egrid(BB * 128), blk256, 0, stream>>>(head, headh, BB * 128);

  cvt_f32_to_f16<<<egrid(NLOC * 128), blk256, 0, stream>>>(ol_W, wh, NLOC * 128);
  gemm_f16_wmma_tiled<<<ggrid(BB, NLOC), blk256, 0, stream>>>(headh, wh, ol_b, out, BB, NLOC, 128);
  add_loc_bias<<<egrid(BB * NLOC), blk256, 0, stream>>>(out, loc_bias, BB * NLOC);
}
